// FeaturePropagation_9998683865705
// MI455X (gfx1250) — matrix operand order repaired and checked by reading, not compiled, not run on hardware
//
#include <hip/hip_runtime.h>
#include <hip/hip_bf16.h>

// ---------------------------------------------------------------------------
// Types for CDNA5 WMMA (wave32): D = A(16x32 bf16) * B(32x16 bf16) + C(16x16 f32)
// ---------------------------------------------------------------------------
typedef __attribute__((ext_vector_type(16))) __bf16 v16bf;
typedef __attribute__((ext_vector_type(8)))  __bf16 v8bf;
typedef __attribute__((ext_vector_type(8)))  float  v8f;

#define N_L     2048
#define N_FINE  65536
#define C_FEAT  256
#define C_ORIG  64
#define IN_CH   323
#define KPAD0   352     // 323 padded up to a multiple of 32

// ---------------------------------------------------------------------------
// Kernel 1: 3-NN inverse-distance^2 interpolation + build padded bf16 X0 row.
// All 2048 coarse points live in LDS (32 KB); one fine point per lane.
// ---------------------------------------------------------------------------
__global__ __launch_bounds__(256)
void interp_build_x0(const float* __restrict__ coords,   // [2048,3]
                     const float* __restrict__ feats,    // [2048,256]
                     const float* __restrict__ q,        // [65536,3]
                     const float* __restrict__ qf,       // [65536,64]
                     __bf16* __restrict__ X0)            // [65536,KPAD0]
{
    __shared__ float4 sC[N_L];
    for (int j = threadIdx.x; j < N_L; j += 256) {
        float cx = coords[j * 3 + 0];
        float cy = coords[j * 3 + 1];
        float cz = coords[j * 3 + 2];
        sC[j] = make_float4(cx, cy, cz, cx * cx + cy * cy + cz * cz);
    }
    __syncthreads();

    const int i = blockIdx.x * 256 + threadIdx.x;
    const float qx = q[i * 3 + 0];
    const float qy = q[i * 3 + 1];
    const float qz = q[i * 3 + 2];
    const float q2 = qx * qx + qy * qy + qz * qz;

    float bd0 = 1e30f, bd1 = 1e30f, bd2 = 1e30f;
    int   bi0 = 0,     bi1 = 0,     bi2 = 0;
    for (int j = 0; j < N_L; ++j) {
        float4 c = sC[j];
        float d2 = q2 + c.w - 2.0f * (qx * c.x + qy * c.y + qz * c.z);
        if (d2 < bd2) {
            if (d2 < bd1) {
                if (d2 < bd0) { bd2 = bd1; bi2 = bi1; bd1 = bd0; bi1 = bi0; bd0 = d2; bi0 = j; }
                else          { bd2 = bd1; bi2 = bi1; bd1 = d2;  bi1 = j; }
            } else            { bd2 = d2;  bi2 = j; }
        }
    }
    float d0 = fmaxf(sqrtf(fmaxf(bd0, 0.0f)), 1e-6f);
    float d1 = fmaxf(sqrtf(fmaxf(bd1, 0.0f)), 1e-6f);
    float d2 = fmaxf(sqrtf(fmaxf(bd2, 0.0f)), 1e-6f);
    float w0 = 1.0f / (d0 * d0);
    float w1 = 1.0f / (d1 * d1);
    float w2 = 1.0f / (d2 * d2);
    float inv = 1.0f / (w0 + w1 + w2);
    w0 *= inv; w1 *= inv; w2 *= inv;

    __bf16* xr = X0 + (size_t)i * KPAD0;
    xr[0] = (__bf16)qx; xr[1] = (__bf16)qy; xr[2] = (__bf16)qz;
    #pragma unroll 4
    for (int c = 0; c < C_ORIG; ++c) xr[3 + c] = (__bf16)qf[i * C_ORIG + c];
    const float* f0 = feats + (size_t)bi0 * C_FEAT;
    const float* f1 = feats + (size_t)bi1 * C_FEAT;
    const float* f2 = feats + (size_t)bi2 * C_FEAT;
    #pragma unroll 4
    for (int c = 0; c < C_FEAT; ++c)
        xr[3 + C_ORIG + c] = (__bf16)(w0 * f0[c] + w1 * f1[c] + w2 * f2[c]);
    for (int c = IN_CH; c < KPAD0; ++c) xr[c] = (__bf16)0.0f;
}

// ---------------------------------------------------------------------------
// Kernel 2: pack f32 weights [O,Kin] into bf16 B-fragment layout.
// For V_WMMA_F32_16X16X32_BF16: lane L holds column n = nt*16 + (L&15),
// 16 contiguous K values K = kt*32 + (L>>4)*16 + e.
// Packed offset = (((nt*nKT + kt)*32 + lane)*16 + e).
// ---------------------------------------------------------------------------
__global__ void pack_w(const float* __restrict__ w, __bf16* __restrict__ wp,
                       int Kin, int Kpad, int O)
{
    int idx = blockIdx.x * blockDim.x + threadIdx.x;
    if (idx >= Kpad * O) return;
    int e    = idx & 15;
    int lane = (idx >> 4) & 31;
    int nKT  = Kpad >> 5;
    int rest = idx >> 9;
    int kt   = rest % nKT;
    int nt   = rest / nKT;
    int n = nt * 16 + (lane & 15);
    int k = kt * 32 + ((lane >> 4) << 4) + e;
    float v = (k < Kin) ? w[(size_t)n * Kin + k] : 0.0f;
    wp[idx] = (__bf16)v;
}

// ---------------------------------------------------------------------------
// Kernel 3: bf16 WMMA GEMM. Y[M,O] = X[M,KPAD](bf16) * Wpacked + bias.
// KPAD/O are compile-time so the whole K-loop fully unrolls: every fragment
// is an SSA value (registers only — no scratch/LDS demotion) and the compiler
// is free to hoist load clauses ahead of the WMMA stream with partial waits.
// Wave -> 16x64 tile (4 v8f accumulators); block (8 waves) -> 128x64.
// Grid: (O/64, M/128). No divergence: EXEC is all-ones for every WMMA.
// ---------------------------------------------------------------------------
template <int KPAD, int O>
__global__ __launch_bounds__(256)
void gemm_bf16(const __bf16* __restrict__ X, const __bf16* __restrict__ Wp,
               const float* __restrict__ bias, float* __restrict__ Y)
{
    constexpr int nKT = KPAD / 32;
    const int lane = threadIdx.x & 31;
    const int wave = threadIdx.x >> 5;
    const int half = lane >> 4;
    const int l16  = lane & 15;
    const int m0   = blockIdx.y * 128 + wave * 16;
    const int n0   = blockIdx.x * 64;

    const __bf16* xrow = X + (size_t)(m0 + l16) * KPAD;
    // Warm this activation row (global_prefetch_b8, cacheline-granular).
    #pragma unroll
    for (int off = 0; off < KPAD; off += 256)
        __builtin_prefetch(xrow + off, 0, 1);

    const __bf16* wbase = Wp + ((size_t)blockIdx.x * 4 * nKT * 32 + lane) * 16;

    v8f acc[4] = {};

    #pragma unroll
    for (int kt = 0; kt < nKT; ++kt) {
        // A fragment (contiguous K, matching packed B): row = l16,
        // elements 0..15 -> K = kt*32 + half*16 + e
        union { v16bf v; v8bf h[2]; } a;
        const __bf16* ap = xrow + kt * 32 + half * 16;
        a.h[0] = *(const v8bf*)(ap);
        a.h[1] = *(const v8bf*)(ap + 8);

        #pragma unroll
        for (int t = 0; t < 4; ++t) {
            const v16bf b = *(const v16bf*)(wbase + (size_t)(t * nKT + kt) * 32 * 16);
            acc[t] = __builtin_amdgcn_wmma_f32_16x16x32_bf16(
                         false, a.v, false, b, (short)0, acc[t], false, false);
        }
    }

    // C/D layout: VGPR r -> row m = half*8 + r, col n = l16.
    #pragma unroll
    for (int t = 0; t < 4; ++t) {
        const int nc = n0 + t * 16 + l16;
        const float bv = bias[nc];
        #pragma unroll
        for (int r = 0; r < 8; ++r) {
            const int mr = m0 + half * 8 + r;
            Y[(size_t)mr * O + nc] = acc[t][r] + bv;
        }
    }
}

// ---------------------------------------------------------------------------
// Kernels 4/5: deterministic two-stage per-column mean/var (O == 256).
// ---------------------------------------------------------------------------
__global__ __launch_bounds__(256)
void col_partial(const float* __restrict__ Y, float* __restrict__ part)
{
    const int t = threadIdx.x;     // column 0..255
    const int b = blockIdx.x;      // row chunk 0..255 (256 rows each)
    float s1 = 0.0f, s2 = 0.0f;
    const float* p = Y + (size_t)b * 256 * 256 + t;
    #pragma unroll 4
    for (int r = 0; r < 256; ++r) {
        float v = p[(size_t)r * 256];
        s1 += v; s2 += v * v;
    }
    part[(size_t)b * 512 + t]       = s1;
    part[(size_t)b * 512 + 256 + t] = s2;
}

__global__ void col_finalize(const float* __restrict__ part,
                             const float* __restrict__ g,
                             const float* __restrict__ be,
                             float* __restrict__ scsh)   // [512]: scale, shift
{
    const int t = threadIdx.x;     // 256 threads
    float s1 = 0.0f, s2 = 0.0f;
    for (int b = 0; b < 256; ++b) {
        s1 += part[b * 512 + t];
        s2 += part[b * 512 + 256 + t];
    }
    const float invN = 1.0f / (float)N_FINE;
    float mean = s1 * invN;
    float var  = s2 * invN - mean * mean;       // biased, like jnp.var
    float sc   = g[t] * rsqrtf(var + 1e-5f);
    scsh[t]       = sc;
    scsh[256 + t] = be[t] - mean * sc;
}

__global__ __launch_bounds__(256)
void bn_relu_bf16(const float* __restrict__ Y, const float* __restrict__ scsh,
                  __bf16* __restrict__ Xn)
{
    const size_t i = (size_t)blockIdx.x * 256 + threadIdx.x;
    const int c = (int)(i & 255);
    float v = Y[i] * scsh[c] + scsh[256 + c];
    Xn[i] = (__bf16)fmaxf(v, 0.0f);
}

// ---------------------------------------------------------------------------
// Host launcher
// ---------------------------------------------------------------------------
extern "C" void kernel_launch(void* const* d_in, const int* in_sizes, int n_in,
                              void* d_out, int out_size, void* d_ws, size_t ws_size,
                              hipStream_t stream)
{
    const float* coords = (const float*)d_in[0];   // [2048,3]
    const float* feats  = (const float*)d_in[1];   // [2048,256]
    const float* oc     = (const float*)d_in[2];   // [65536,3]
    const float* of     = (const float*)d_in[3];   // [65536,64]
    const float* w0 = (const float*)d_in[4];
    const float* b0 = (const float*)d_in[5];
    const float* g0 = (const float*)d_in[6];
    const float* be0 = (const float*)d_in[7];
    const float* w1 = (const float*)d_in[8];
    const float* b1 = (const float*)d_in[9];
    const float* g1 = (const float*)d_in[10];
    const float* be1 = (const float*)d_in[11];
    const float* w2 = (const float*)d_in[12];
    const float* b2 = (const float*)d_in[13];

    char* ws = (char*)d_ws;
    // X region (X0 bf16 [65536,352]; later reused for X1/X2 bf16 [65536,256])
    __bf16* X0  = (__bf16*)(ws);
    __bf16* Xn  = (__bf16*)(ws);                           // reuse after X0 dead
    size_t off  = (size_t)N_FINE * KPAD0 * sizeof(__bf16); // 46,137,344
    float*  Yb  = (float*)(ws + off);                      // [65536,256] f32
    off += (size_t)N_FINE * 256 * sizeof(float);           // +67,108,864
    __bf16* Wp0 = (__bf16*)(ws + off); off += (size_t)KPAD0 * 256 * sizeof(__bf16);
    __bf16* Wp1 = (__bf16*)(ws + off); off += (size_t)256 * 256 * sizeof(__bf16);
    __bf16* Wp2 = (__bf16*)(ws + off); off += (size_t)256 * 128 * sizeof(__bf16);
    float*  part = (float*)(ws + off); off += (size_t)256 * 512 * sizeof(float);
    float*  scsh = (float*)(ws + off); off += 512 * sizeof(float);
    (void)ws_size; (void)in_sizes; (void)n_in; (void)out_size;

    // Pack weights into WMMA B-fragment layout.
    pack_w<<<(KPAD0 * 256 + 255) / 256, 256, 0, stream>>>(w0, Wp0, IN_CH, KPAD0, 256);
    pack_w<<<(256 * 256 + 255) / 256,  256, 0, stream>>>(w1, Wp1, 256, 256, 256);
    pack_w<<<(256 * 128 + 255) / 256,  256, 0, stream>>>(w2, Wp2, 256, 256, 128);

    // KNN interpolation + concat -> X0 (bf16, padded).
    interp_build_x0<<<N_FINE / 256, 256, 0, stream>>>(coords, feats, oc, of, X0);

    // Layer 0: GEMM -> BN stats -> BN+ReLU -> X1 (bf16).
    gemm_bf16<KPAD0, 256><<<dim3(256 / 64, N_FINE / 128), 256, 0, stream>>>(X0, Wp0, b0, Yb);
    col_partial<<<256, 256, 0, stream>>>(Yb, part);
    col_finalize<<<1, 256, 0, stream>>>(part, g0, be0, scsh);
    bn_relu_bf16<<<N_FINE, 256, 0, stream>>>(Yb, scsh, Xn);

    // Layer 1: GEMM -> BN stats -> BN+ReLU -> X2 (bf16, same buffer).
    gemm_bf16<256, 256><<<dim3(256 / 64, N_FINE / 128), 256, 0, stream>>>(Xn, Wp1, b1, Yb);
    col_partial<<<256, 256, 0, stream>>>(Yb, part);
    col_finalize<<<1, 256, 0, stream>>>(part, g1, be1, scsh);
    bn_relu_bf16<<<N_FINE, 256, 0, stream>>>(Yb, scsh, Xn);

    // Layer 2: plain Linear straight into d_out (f32 [65536,128]).
    gemm_bf16<256, 128><<<dim3(128 / 64, N_FINE / 128), 256, 0, stream>>>(Xn, Wp2, b2, (float*)d_out);
}
